// SMOKEPredictor_45311904972836
// MI455X (gfx1250) — matrix operand order repaired and checked by reading, not compiled, not run on hardware
//
#include <hip/hip_runtime.h>
#include <hip/hip_bf16.h>

typedef __attribute__((ext_vector_type(16))) __bf16 v16bf;
typedef __attribute__((ext_vector_type(8)))  float  v8f;

#define BN_EPS 1e-5f
#define HM_EPS 1e-4f
#define W4 480
#define H4 160
#define NPIX 76800          // 160*480
#define NCLS 3
#define HC 256
#define NKT 18              // 576/32 K-chunks
#define PSTRIDE 72          // bf16 elements per (row,col) entry: mult of 8 -> 16B aligned slices
#define NEG_BIG -1.0e30f

// ---------------- helpers ----------------
__device__ inline unsigned short f32_to_bf16(float f) {
    unsigned int u = __builtin_bit_cast(unsigned int, f);
    unsigned int r = u + 0x7FFFu + ((u >> 16) & 1u);   // round-to-nearest-even
    return (unsigned short)(r >> 16);
}
__device__ inline float sigmoidf(float x) { return 1.f / (1.f + __expf(-x)); }

// A-fragment K index for element e (=2*vgpr+half), half-wave h
__device__ inline int a_k_of(int e, int h) { return 16 * h + e; }

// ---------------- kernel 0: pack cls_w1 into WMMA A-fragment order (bf16 pairs) ----------------
// Apack[(m*18+kt)*32*8 + lane*8 + v] : dword = {bf16(K even) lo, bf16(K odd) hi}
__global__ void pack_w_kernel(const float* __restrict__ w, unsigned int* __restrict__ Apack) {
    int d = blockIdx.x * 256 + threadIdx.x;
    if (d >= 16 * NKT * 32 * 8) return;
    int v    = d & 7;
    int lane = (d >> 3) & 31;
    int kt   = (d >> 8) % NKT;
    int m    = d / (NKT * 256);
    int h    = lane >> 4;
    unsigned int out = 0;
    for (int hh = 0; hh < 2; ++hh) {
        int e  = 2 * v + hh;
        int Kg = kt * 32 + a_k_of(e, h);
        int tap = Kg >> 6, ic = Kg & 63;
        int ky = tap / 3, kx = tap % 3;
        int o  = m * 16 + (lane & 15);
        float f = w[((o * 64 + ic) * 3 + ky) * 3 + kx];
        out |= ((unsigned int)f32_to_bf16(f)) << (16 * hh);
    }
    Apack[d] = out;
}

// ---------------- kernel 1: fused conv3x3(64->256)+BN+ReLU + conv1x1(256->3)+sigmoid+clip ----------------
// grid: 4800 blocks (one per 16-wide spatial strip), block: 512 threads = 16 waves.
__global__ __launch_bounds__(512) void cls_head_kernel(
    const float* __restrict__ up4, const unsigned int* __restrict__ Apack,
    const float* __restrict__ b1, const float* __restrict__ bg, const float* __restrict__ bb,
    const float* __restrict__ bm, const float* __restrict__ bv,
    const float* __restrict__ w2, const float* __restrict__ b2,
    float* __restrict__ heat)
{
    // input patch pre-converted to bf16: [r][col][ic], ic contiguous, stride 72 (16B-aligned slices)
    __shared__ __align__(16) unsigned short sPatch[3 * 18 * PSTRIDE];
    __shared__ float sX[HC * 16];           // relu'ed 256xN tile for the 1x1 conv

    const int tid = threadIdx.x;
    const int tb  = blockIdx.x;
    const int y   = tb / 30;
    const int xt  = (tb % 30) * 16;

    // stage input patch (64 ch x 3 rows x 18 cols), convert f32->bf16 once, zero-pad borders
    for (int i = tid; i < 64 * 54; i += 512) {
        int ic = i / 54, rem = i % 54;
        int r = rem / 18, col = rem % 18;
        int gy = y + r - 1, gx = xt + col - 1;
        float v = 0.f;
        if (gy >= 0 && gy < H4 && gx >= 0 && gx < W4)
            v = up4[ic * NPIX + gy * W4 + gx];
        sPatch[(r * 18 + col) * PSTRIDE + ic] = f32_to_bf16(v);
    }
    __syncthreads();

    const int wave = tid >> 5;       // M-tile: out channels wave*16 .. +15
    const int lane = tid & 31;
    const int n    = lane & 15;      // N (position in strip)
    const int h    = lane >> 4;      // half-wave
    const int c0   = wave * 16;

    v8f acc = {0.f, 0.f, 0.f, 0.f, 0.f, 0.f, 0.f, 0.f};
    const uint4* apBase = (const uint4*)(Apack + (size_t)(wave * NKT) * 256 + (size_t)lane * 8);

#pragma unroll
    for (int kt = 0; kt < NKT; ++kt) {
        // A fragment: two global b128 loads of pre-packed bf16 weight pairs
        union { uint4 q[2]; v16bf v; } af;
        const uint4* ap = apBase + (size_t)kt * 64;   // 32 lanes * 8 dwords = 64 uint4 per chunk
        af.q[0] = ap[0];
        af.q[1] = ap[1];

        // B fragment: 32 contiguous bf16 from LDS (two ds_load_b128), already pair-ordered
        const int tap = kt >> 1;
        const int ky  = tap / 3, kx = tap % 3;
        const int ic0 = (kt & 1) * 32;
        union { uint4 q[2]; v16bf v; } bfr;
        const uint4* bp = (const uint4*)(&sPatch[(ky * 18 + (n + kx)) * PSTRIDE + ic0 + 16 * h]);
        bfr.q[0] = bp[0];
        bfr.q[1] = bp[1];

        acc = __builtin_amdgcn_wmma_f32_16x16x32_bf16(false, af.v, false, bfr.v,
                                                      (short)0, acc, false, false);
    }

    // BN + ReLU per channel, stash tile to LDS for the 1x1 conv
#pragma unroll
    for (int j = 0; j < 8; ++j) {
        int ch = c0 + j + 8 * h;
        float scale = bg[ch] * rsqrtf(bv[ch] + BN_EPS);
        float val = (acc[j] + b1[ch] - bm[ch]) * scale + bb[ch];
        sX[ch * 16 + n] = fmaxf(val, 0.f);
    }
    __syncthreads();

    // 1x1 conv 256->3 + sigmoid + clip
    if (tid < 48) {
        int cl = tid / 16, nn = tid % 16;
        float a = b2[cl];
        for (int ch = 0; ch < HC; ++ch) a += sX[ch * 16 + nn] * w2[cl * HC + ch];
        float s = sigmoidf(a);
        s = fminf(fmaxf(s, HM_EPS), 1.f - HM_EPS);
        heat[cl * NPIX + y * W4 + xt + nn] = s;
    }
}

// ---------------- kernel 2: 3x3 NMS ----------------
__global__ void nms_kernel(const float* __restrict__ heat, float* __restrict__ out) {
    int i = blockIdx.x * 256 + threadIdx.x;
    if (i >= NCLS * NPIX) return;
    int c = i / NPIX, p = i % NPIX;
    int y = p / W4, x = p % W4;
    float v = heat[i];
    float mx = NEG_BIG;
    for (int dy = -1; dy <= 1; ++dy) {
        int ny = y + dy;
        if (ny < 0 || ny >= H4) continue;
        for (int dx = -1; dx <= 1; ++dx) {
            int nx = x + dx;
            if (nx < 0 || nx >= W4) continue;
            mx = fmaxf(mx, heat[c * NPIX + ny * W4 + nx]);
        }
    }
    out[i] = (mx == v) ? v : 0.f;
}

// ---------------- kernel 3: per-class iterative top-50 (argmax + suppress) ----------------
__global__ __launch_bounds__(512) void topk_class_kernel(float* __restrict__ hn,
                                                         float* __restrict__ candS,
                                                         int* __restrict__ candI) {
    __shared__ float sv[512];
    __shared__ int   si[512];
    const int c = blockIdx.x, tid = threadIdx.x;
    float* h = hn + c * NPIX;
    for (int it = 0; it < 50; ++it) {
        float best = NEG_BIG; int bi = NPIX;
        for (int i = tid; i < NPIX; i += 512) {
            float v = h[i];
            if (v > best || (v == best && i < bi)) { best = v; bi = i; }
        }
        sv[tid] = best; si[tid] = bi;
        __syncthreads();
        for (int s = 256; s > 0; s >>= 1) {
            if (tid < s) {
                float ov = sv[tid + s]; int oi = si[tid + s];
                if (ov > sv[tid] || (ov == sv[tid] && oi < si[tid])) { sv[tid] = ov; si[tid] = oi; }
            }
            __syncthreads();
        }
        if (tid == 0) {
            candS[c * 50 + it] = sv[0];
            candI[c * 50 + it] = si[0];
            h[si[0]] = NEG_BIG;     // suppress (buffer rebuilt by nms_kernel each call)
        }
        __syncthreads();
    }
}

// ---------------- kernel 4: merge 150 -> top 50, emit scores/cls/ys/xs outputs ----------------
__global__ void merge_topk_kernel(const float* __restrict__ candS, const int* __restrict__ candI,
                                  float* __restrict__ out, int* __restrict__ selX, int* __restrict__ selY) {
    __shared__ float cs[150];
    __shared__ int   ci[150];
    __shared__ int   used[150];
    int tid = threadIdx.x;
    if (tid < 150) { cs[tid] = candS[tid]; ci[tid] = candI[tid]; used[tid] = 0; }
    __syncthreads();
    if (tid == 0) {
        for (int k = 0; k < 50; ++k) {
            float best = NEG_BIG; int bi = 0;
            for (int i = 0; i < 150; ++i)
                if (!used[i] && cs[i] > best) { best = cs[i]; bi = i; }
            used[bi] = 1;
            int flat = ci[bi];
            int yv = flat / W4, xv = flat % W4;
            out[400 + k] = best;               // scores
            out[450 + k] = (float)(bi / 50);   // clses
            out[500 + k] = (float)yv;          // ys
            out[550 + k] = (float)xv;          // xs
            selX[k] = xv; selY[k] = yv;
        }
    }
}

// ---------------- kernel 5: reg conv at 50 points + feature gather + 8x640 head + postprocess ----------------
__global__ __launch_bounds__(256) void reg_head_kernel(
    const float* __restrict__ up4, const float* __restrict__ up8, const float* __restrict__ up16,
    const float* __restrict__ w1, const float* __restrict__ b1,
    const float* __restrict__ bg, const float* __restrict__ bb,
    const float* __restrict__ bm, const float* __restrict__ bv,
    const float* __restrict__ w3, const float* __restrict__ b3,
    const int* __restrict__ selX, const int* __restrict__ selY, float* __restrict__ out)
{
    __shared__ float sP[576];     // 64ch x 3x3 patch
    __shared__ float sPoi[640];
    __shared__ float sReg[8];
    const int k = blockIdx.x, tid = threadIdx.x;
    const int x = selX[k], y = selY[k];

    for (int i = tid; i < 576; i += 256) {
        int ic = i / 9, rem = i % 9;
        int ky = rem / 3, kx = rem % 3;
        int gy = y + ky - 1, gx = x + kx - 1;
        float v = 0.f;
        if (gy >= 0 && gy < H4 && gx >= 0 && gx < W4)
            v = up4[ic * NPIX + gy * W4 + gx];
        sP[i] = v;
    }
    __syncthreads();

    {   // reg conv channel tid (0..255) at this point + BN + ReLU
        float a = b1[tid];
        for (int ic = 0; ic < 64; ++ic) {
            const float* wr = &w1[(tid * 64 + ic) * 9];
            const float* pr = &sP[ic * 9];
#pragma unroll
            for (int t = 0; t < 9; ++t) a += pr[t] * wr[t];
        }
        float scale = bg[tid] * rsqrtf(bv[tid] + BN_EPS);
        float val = (a - bm[tid]) * scale + bb[tid];
        sPoi[tid] = fmaxf(val, 0.f);
    }
    if (tid < 128) sPoi[256 + tid] = up8[tid * 19200 + (y >> 1) * 240 + (x >> 1)];
    sPoi[384 + tid] = up16[tid * 4800 + (y >> 2) * 120 + (x >> 2)];
    __syncthreads();

    if (tid < 8) {
        float a = b3[tid];
        for (int c = 0; c < 640; ++c) a += w3[tid * 640 + c] * sPoi[c];
        sReg[tid] = a;
    }
    __syncthreads();
    if (tid < 8) {
        float v = sReg[tid];
        if (tid >= 3 && tid < 6) v = sigmoidf(v) - 0.5f;
        else if (tid >= 6) {
            float nrm = fmaxf(sqrtf(sReg[6] * sReg[6] + sReg[7] * sReg[7]), 1e-12f);
            v = v / nrm;
        }
        out[tid * 50 + k] = v;   // head_regression (1,8,50)
    }
}

// ---------------- launch ----------------
extern "C" void kernel_launch(void* const* d_in, const int* in_sizes, int n_in,
                              void* d_out, int out_size, void* d_ws, size_t ws_size,
                              hipStream_t stream) {
    const float* up16   = (const float*)d_in[0];
    const float* up8    = (const float*)d_in[1];
    const float* up4    = (const float*)d_in[2];
    const float* cls_w1 = (const float*)d_in[3];
    const float* cls_b1 = (const float*)d_in[4];
    const float* cls_g  = (const float*)d_in[5];
    const float* cls_bb = (const float*)d_in[6];
    const float* cls_m  = (const float*)d_in[7];
    const float* cls_v  = (const float*)d_in[8];
    const float* cls_w2 = (const float*)d_in[9];
    const float* cls_b2 = (const float*)d_in[10];
    const float* reg_w1 = (const float*)d_in[11];
    const float* reg_b1 = (const float*)d_in[12];
    const float* reg_g  = (const float*)d_in[13];
    const float* reg_bb = (const float*)d_in[14];
    const float* reg_m  = (const float*)d_in[15];
    const float* reg_v  = (const float*)d_in[16];
    const float* reg3_w = (const float*)d_in[17];
    const float* reg3_b = (const float*)d_in[18];
    float* out = (float*)d_out;

    char* ws = (char*)d_ws;
    unsigned int* Apack = (unsigned int*)(ws + 0);        // 294912 B
    float* heat   = (float*)(ws + 294912);                // 921600 B
    float* hnms   = (float*)(ws + 1216512);               // 921600 B
    float* candS  = (float*)(ws + 2138112);               // 600 B
    int*   candI  = (int*)  (ws + 2138712);               // 600 B
    int*   selX   = (int*)  (ws + 2139312);               // 200 B
    int*   selY   = (int*)  (ws + 2139512);               // 200 B

    pack_w_kernel<<<dim3(288), dim3(256), 0, stream>>>(cls_w1, Apack);
    cls_head_kernel<<<dim3(4800), dim3(512), 0, stream>>>(
        up4, Apack, cls_b1, cls_g, cls_bb, cls_m, cls_v, cls_w2, cls_b2, heat);
    nms_kernel<<<dim3(900), dim3(256), 0, stream>>>(heat, hnms);
    topk_class_kernel<<<dim3(3), dim3(512), 0, stream>>>(hnms, candS, candI);
    merge_topk_kernel<<<dim3(1), dim3(160), 0, stream>>>(candS, candI, out, selX, selY);
    reg_head_kernel<<<dim3(50), dim3(256), 0, stream>>>(
        up4, up8, up16, reg_w1, reg_b1, reg_g, reg_bb, reg_m, reg_v,
        reg3_w, reg3_b, selX, selY, out);
}
